// MixSelfAttention_88046829568161
// MI455X (gfx1250) — matrix operand order repaired and checked by reading, not compiled, not run on hardware
//
#include <hip/hip_runtime.h>
#include <hip/hip_bf16.h>

typedef __attribute__((ext_vector_type(16))) __bf16 v16bf;
typedef __attribute__((ext_vector_type(8)))  float  v8f;

#define SEQ   1024
#define EDIM  64
#define NHEAD 8
#define NBH   64          // B*H = 8*8
#define NMODE 64
#define MIXSCALE (1.0f / (512.0f * 512.0f))

// ---------------------------------------------------------------------------
// Kernel 1: circular autocorrelation per (b,h,e); bf16 output [bh][lag][e]
// ---------------------------------------------------------------------------
__global__ __launch_bounds__(256)
void autocorr_kernel(const float* __restrict__ q, __bf16* __restrict__ corrB) {
    __shared__ float xs[SEQ];
    __shared__ float red[256];
    const int blk = blockIdx.x;            // NBH * EDIM
    const int bh = blk >> 6, e = blk & 63;
    const int b = bh >> 3, h = bh & 7;
    const int tid = threadIdx.x;

    #pragma unroll
    for (int i = 0; i < 4; ++i) {
        int l = tid + i * 256;
        xs[l] = q[(((size_t)b * SEQ + l) * NHEAD + h) * EDIM + e];
    }
    __syncthreads();

    float s = 0.f;
    #pragma unroll
    for (int i = 0; i < 4; ++i) s += xs[tid + i * 256];
    red[tid] = s; __syncthreads();
    for (int st = 128; st > 0; st >>= 1) { if (tid < st) red[tid] += red[tid + st]; __syncthreads(); }
    const float mean = red[0] * (1.0f / SEQ);
    __syncthreads();

    #pragma unroll
    for (int i = 0; i < 4; ++i) xs[tid + i * 256] -= mean;
    __syncthreads();

    float d = 0.f;
    #pragma unroll
    for (int i = 0; i < 4; ++i) { float v = xs[tid + i * 256]; d += v * v; }
    red[tid] = d; __syncthreads();
    for (int st = 128; st > 0; st >>= 1) { if (tid < st) red[tid] += red[tid + st]; __syncthreads(); }
    const float inv = 1.0f / red[0];

    #pragma unroll
    for (int i = 0; i < 4; ++i) {
        int lag = tid + i * 256;
        float num = 0.f;
        for (int t = 0; t < SEQ; ++t) num += xs[t] * xs[(t + lag) & (SEQ - 1)];
        corrB[((size_t)bh * SEQ + lag) * EDIM + e] = (__bf16)(num * inv);
    }
}

// ---------------------------------------------------------------------------
// Kernel 1b: pack tf_queries [B][L][H][E] f32 -> [bh][l][e] bf16
// ---------------------------------------------------------------------------
__global__ __launch_bounds__(256)
void pack_tf_kernel(const float* __restrict__ tfq, __bf16* __restrict__ tfB) {
    #pragma unroll
    for (int i = 0; i < 4; ++i) {
        size_t idx = (size_t)blockIdx.x * 1024 + (size_t)i * 256 + threadIdx.x;
        int e = idx & 63;
        int l = (idx >> 6) & (SEQ - 1);
        int bh = (int)(idx >> 16);
        int b = bh >> 3, h = bh & 7;
        tfB[idx] = (__bf16)tfq[(((size_t)b * SEQ + l) * NHEAD + h) * EDIM + e];
    }
}

// ---------------------------------------------------------------------------
// Kernel 1c: pack values [B][L][H][E] f32 -> transposed [bh][d][s] bf16
// ---------------------------------------------------------------------------
__global__ __launch_bounds__(256)
void pack_vals_kernel(const float* __restrict__ vals, __bf16* __restrict__ valsT) {
    #pragma unroll
    for (int i = 0; i < 4; ++i) {
        size_t idx = (size_t)blockIdx.x * 1024 + (size_t)i * 256 + threadIdx.x;
        int s = idx & (SEQ - 1);
        int d = (idx >> 10) & 63;
        int bh = (int)(idx >> 16);
        int b = bh >> 3, h = bh & 7;
        valsT[idx] = (__bf16)vals[(((size_t)b * SEQ + s) * NHEAD + h) * EDIM + d];
    }
}

// ---------------------------------------------------------------------------
// Kernel 2: 64-mode DFT of raw x per (b,h,e); output [bh][m][e] complex
// ---------------------------------------------------------------------------
__global__ __launch_bounds__(256)
void dft_kernel(const float* __restrict__ q, float* __restrict__ xft) {
    __shared__ float xs[SEQ];
    __shared__ float redr[256], redi[256];
    const int blk = blockIdx.x;            // NBH * EDIM
    const int bh = blk >> 6, e = blk & 63;
    const int b = bh >> 3, h = bh & 7;
    const int tid = threadIdx.x;

    #pragma unroll
    for (int i = 0; i < 4; ++i) {
        int l = tid + i * 256;
        xs[l] = q[(((size_t)b * SEQ + l) * NHEAD + h) * EDIM + e];
    }
    __syncthreads();

    const int m = tid & 63, part = tid >> 6;
    float sr = 0.f, si = 0.f;
    for (int t = part * 256; t < part * 256 + 256; ++t) {
        float th = -6.283185307179586f * (float)((m * t) & (SEQ - 1)) * (1.0f / SEQ);
        float sn, cs; __sincosf(th, &sn, &cs);
        sr += xs[t] * cs; si += xs[t] * sn;
    }
    redr[tid] = sr; redi[tid] = si;
    __syncthreads();
    if (part == 0) {
        float ar = redr[m] + redr[m + 64] + redr[m + 128] + redr[m + 192];
        float ai = redi[m] + redi[m + 64] + redi[m + 128] + redi[m + 192];
        float* p = xft + (((size_t)bh * NMODE + m) * EDIM + e) * 2;
        p[0] = ar; p[1] = ai;
    }
}

// ---------------------------------------------------------------------------
// Kernel 3: per-mode complex mix  y[bh,o,m] = sum_i xft[bh,m,i] * w1[h,i,o,m]
// ---------------------------------------------------------------------------
__global__ __launch_bounds__(64)
void modemix_kernel(const float* __restrict__ xft, const float* __restrict__ w1r,
                    const float* __restrict__ w1i, float* __restrict__ y) {
    __shared__ float xr[EDIM], xi[EDIM];
    const int blk = blockIdx.x;            // NBH * NMODE
    const int bh = blk >> 6, m = blk & 63;
    const int h = bh & 7;
    const int o = threadIdx.x;

    const float* p = xft + ((size_t)bh * NMODE + m) * EDIM * 2;
    xr[o] = p[2 * o]; xi[o] = p[2 * o + 1];
    __syncthreads();

    float ar = 0.f, ai = 0.f;
    for (int i = 0; i < EDIM; ++i) {
        float wr = w1r[(((size_t)h * EDIM + i) * EDIM + o) * NMODE + m];
        float wi = w1i[(((size_t)h * EDIM + i) * EDIM + o) * NMODE + m];
        ar += xr[i] * wr - xi[i] * wi;
        ai += xr[i] * wi + xi[i] * wr;
    }
    float* qy = y + (((size_t)bh * EDIM + o) * NMODE + m) * 2;
    qy[0] = ar; qy[1] = ai;
}

// ---------------------------------------------------------------------------
// Kernel 4: fused Gram + mix + softmax + A*V  (WMMA bf16, f32 accumulate)
//   grid: NBH * 16 row-blocks of 64 rows.  8 waves = 4 row tiles x 2 col halves
//   values tiles staged to LDS via CDNA5 async global->LDS loads (ASYNCcnt)
// ---------------------------------------------------------------------------
__global__ __launch_bounds__(256)
void attn_kernel(const __bf16* __restrict__ corrB,   // [bh][l][e] bf16
                 const __bf16* __restrict__ tfB,     // [bh][l][e] bf16
                 const __bf16* __restrict__ valsT,   // [bh][d][s] bf16
                 const float* __restrict__ tw, const float* __restrict__ fw,
                 float* __restrict__ out) {          // [bh][d][l]
    __shared__ float ntw_s[64], nfw_s[64];
    __shared__ __bf16 stage[8][16 * 32];             // per-wave P re-layout buffer
    __shared__ __bf16 vstage[2][64 * 32];            // per-col-half values tile (async)
    __shared__ float vbuf[4][16 * 64];               // col-half combine
    __shared__ float rsbuf[4][16];

    const int blk = blockIdx.x;                      // NBH * 16
    const int bh = blk >> 4;
    const int l0 = (blk & 15) * 64;

    const int tid = threadIdx.x;
    const int wv = tid >> 5, lane = tid & 31;
    const int r = wv >> 1;                           // row tile 0..3
    const int c = wv & 1;                            // column half 0/1
    const int row16 = lane & 15;
    const int hi = lane >> 4;                        // lane group 0/1

    if (tid < 64) {
        float t = tw[l0 + tid], f = fw[l0 + tid];
        float tot = t + f; tot += (tot == 0.0f) ? 1e-8f : 0.0f;
        ntw_s[tid] = t / tot; nfw_s[tid] = f / tot;
    }
    __syncthreads();

    // A fragments (16x32 bf16, contiguous k = 16*hi + s layout) — loaded once per block
    const int lrow = l0 + r * 16 + row16;
    const __bf16* crow = corrB + ((size_t)bh * SEQ + lrow) * EDIM;
    const __bf16* trow = tfB + ((size_t)bh * SEQ + lrow) * EDIM;
    v16bf a_c[2], a_t[2];
    #pragma unroll
    for (int ch = 0; ch < 2; ++ch)
        #pragma unroll
        for (int hh = 0; hh < 16; ++hh) {
            int k = ch * 32 + hi * 16 + hh;
            a_c[ch][hh] = crow[k];
            a_t[ch][hh] = trow[k];
        }

    v8f vacc[4] = {};
    float rsum[8];
    #pragma unroll
    for (int i = 0; i < 8; ++i) rsum[i] = 0.f;

    float ntw_r[8], nfw_r[8];
    #pragma unroll
    for (int v = 0; v < 8; ++v) {
        int rr = r * 16 + v + hi * 8;
        ntw_r[v] = ntw_s[rr]; nfw_r[v] = nfw_s[rr];
    }

    __bf16* myStage = &stage[wv][0];

    for (int it = 0; it < 32; ++it) {
        const int s0 = c * 512 + it * 16;
        const int scol = s0 + row16;

        // Even iteration: one wave per column half issues async global->LDS
        // loads of the values tile (32 s  x 64 d bf16 = 4 KB) for this pair.
        if (!(it & 1) && r == 0) {
            #pragma unroll
            for (int j = 0; j < 8; ++j) {
                int seg = j * 32 + lane;
                int d = seg >> 2, part = seg & 3;
                const __bf16* gsrc = valsT + ((size_t)bh * EDIM + d) * SEQ + s0 + part * 8;
                unsigned ldsoff = (unsigned)(uintptr_t)(&vstage[c][d * 32 + part * 8]);
                asm volatile("global_load_async_to_lds_b128 %0, %1, off"
                             :: "v"(ldsoff), "v"(gsrc) : "memory");
            }
        }

        // Gram tiles: B fragments are contiguous 16xbf16 vector loads
        const __bf16* ccol = corrB + ((size_t)bh * SEQ + scol) * EDIM;
        const __bf16* tcol = tfB + ((size_t)bh * SEQ + scol) * EDIM;
        v8f accS = {}, accT = {};
        #pragma unroll
        for (int ch = 0; ch < 2; ++ch) {
            v16bf b_c = *(const v16bf*)(ccol + ch * 32 + hi * 16);
            v16bf b_t = *(const v16bf*)(tcol + ch * 32 + hi * 16);
            accS = __builtin_amdgcn_wmma_f32_16x16x32_bf16(false, a_c[ch], false, b_c,
                                                           (short)0, accS, false, false);
            accT = __builtin_amdgcn_wmma_f32_16x16x32_bf16(false, a_t[ch], false, b_t,
                                                           (short)0, accT, false, false);
        }

        // mix + exp (SCALE makes logits tiny: no max subtraction needed) + stage P
        const int colslot = (it & 1) * 16;
        #pragma unroll
        for (int v = 0; v < 8; ++v) {
            float mx = ntw_r[v] * accS[v] + nfw_r[v] * accT[v];
            float p = __expf(MIXSCALE * mx);
            rsum[v] += p;
            myStage[(v + hi * 8) * 32 + colslot + row16] = (__bf16)p;
        }

        // Issuing wave drains its async counter before the group barrier so the
        // values tile is LDS-visible to all waves of this column half.
        if (!(it & 1) && r == 0)
            asm volatile("s_wait_asynccnt 0x0" ::: "memory");
        __syncthreads();

        if (it & 1) {
            // reload P as a 16x32 A-fragment (K = s within the 32-col pair)
            v16bf a_p;
            #pragma unroll
            for (int hh = 0; hh < 16; ++hh) {
                int k = hi * 16 + hh;
                a_p[hh] = myStage[row16 * 32 + k];
            }
            #pragma unroll
            for (int n = 0; n < 4; ++n) {
                int d = n * 16 + row16;
                v16bf b_v = *(const v16bf*)(&vstage[c][d * 32 + hi * 16]);
                vacc[n] = __builtin_amdgcn_wmma_f32_16x16x32_bf16(false, a_p, false, b_v,
                                                                  (short)0, vacc[n], false, false);
            }
        }
        __syncthreads();
    }

    // row sums: butterfly within the 16-lane groups
    #pragma unroll
    for (int off = 1; off < 16; off <<= 1)
        #pragma unroll
        for (int v = 0; v < 8; ++v)
            rsum[v] += __shfl_xor(rsum[v], off, 32);

    if (c == 0) {
        #pragma unroll
        for (int n = 0; n < 4; ++n)
            #pragma unroll
            for (int v = 0; v < 8; ++v)
                vbuf[r][(v + hi * 8) * 64 + n * 16 + row16] = vacc[n][v];
        if (row16 == 0)
            #pragma unroll
            for (int v = 0; v < 8; ++v) rsbuf[r][v + hi * 8] = rsum[v];
    }
    __syncthreads();
    if (c == 1) {
        #pragma unroll
        for (int n = 0; n < 4; ++n)
            #pragma unroll
            for (int v = 0; v < 8; ++v) {
                int prow = v + hi * 8;
                float tot = vacc[n][v] + vbuf[r][prow * 64 + n * 16 + row16];
                float rs = rsum[v] + rsbuf[r][prow];
                int l = l0 + r * 16 + prow;
                int d = n * 16 + row16;
                out[((size_t)bh * EDIM + d) * SEQ + l] = 0.5f * tot / rs;
            }
    }
}

// ---------------------------------------------------------------------------
// Kernel 5: 64-mode inverse transform, accumulate 0.5*x_freq into d_out
// ---------------------------------------------------------------------------
__global__ __launch_bounds__(256)
void irfft_add_kernel(const float* __restrict__ y, float* __restrict__ out) {
    __shared__ float yr[NMODE], yi[NMODE];
    const int blk = blockIdx.x;            // NBH * EDIM
    const int bh = blk >> 6, o = blk & 63;
    const int tid = threadIdx.x;
    if (tid < NMODE) {
        const float* p = y + (((size_t)bh * EDIM + o) * NMODE + tid) * 2;
        yr[tid] = p[0]; yi[tid] = p[1];
    }
    __syncthreads();

    #pragma unroll
    for (int i = 0; i < 4; ++i) {
        int t = tid + i * 256;
        float acc = yr[0];                 // irfft keeps only Re(X0)
        for (int m = 1; m < NMODE; ++m) {
            float th = 6.283185307179586f * (float)((m * t) & (SEQ - 1)) * (1.0f / SEQ);
            float sn, cs; __sincosf(th, &sn, &cs);
            acc += 2.0f * (yr[m] * cs - yi[m] * sn);
        }
        float xf = acc * (1.0f / SEQ);
        size_t idx = ((size_t)bh * EDIM + o) * SEQ + t;
        out[idx] += 0.5f * xf;
    }
}

// ---------------------------------------------------------------------------
extern "C" void kernel_launch(void* const* d_in, const int* in_sizes, int n_in,
                              void* d_out, int out_size, void* d_ws, size_t ws_size,
                              hipStream_t stream) {
    const float* tfq  = (const float*)d_in[0];
    const float* q    = (const float*)d_in[1];
    // d_in[2] = keys (unused by reference), d_in[4] = mask (all true, unused)
    const float* vals = (const float*)d_in[3];
    const float* w1r  = (const float*)d_in[5];
    const float* w1i  = (const float*)d_in[6];
    const float* tw   = (const float*)d_in[7];
    const float* fw   = (const float*)d_in[8];
    float* out = (float*)d_out;

    const size_t NELEM = (size_t)NBH * SEQ * EDIM;   // 4,194,304
    char* ws = (char*)d_ws;
    __bf16* corrB = (__bf16*)ws;                         // 8 MB
    __bf16* tfB   = (__bf16*)(ws + NELEM * 2);           // 8 MB
    __bf16* valsT = (__bf16*)(ws + NELEM * 4);           // 8 MB
    float*  xft   = (float*)(ws + NELEM * 6);            // 2 MB
    float*  ymix  = (float*)(ws + NELEM * 6 + (size_t)NBH * NMODE * EDIM * 2 * 4);

    autocorr_kernel<<<dim3(NBH * EDIM), dim3(256), 0, stream>>>(q, corrB);
    pack_tf_kernel<<<dim3(NELEM / 1024), dim3(256), 0, stream>>>(tfq, tfB);
    pack_vals_kernel<<<dim3(NELEM / 1024), dim3(256), 0, stream>>>(vals, valsT);
    dft_kernel<<<dim3(NBH * EDIM), dim3(256), 0, stream>>>(q, xft);
    modemix_kernel<<<dim3(NBH * NMODE), dim3(64), 0, stream>>>(xft, w1r, w1i, ymix);
    attn_kernel<<<dim3(NBH * 16), dim3(256), 0, stream>>>(corrB, tfB, valsT, tw, fw, out);
    irfft_add_kernel<<<dim3(NBH * EDIM), dim3(256), 0, stream>>>(ymix, out);
}
